// GlobalAttentionGeneral_17867063951923
// MI455X (gfx1250) — matrix operand order repaired and checked by reading, not compiled, not run on hardware
//
#include <hip/hip_runtime.h>

// ---------------------------------------------------------------------------
// Fused AttnGAN GlobalAttentionGeneral for gfx1250 (MI455X), wave32 + WMMA.
// B=32, IDF=128, CDF=256, IH=IW=128 (queryL=16384), SL=24 (padded to 32).
// ---------------------------------------------------------------------------

typedef __bf16 bf16_t;
typedef __attribute__((ext_vector_type(16))) __bf16 v16bf;
typedef __attribute__((ext_vector_type(8)))  __bf16 v8bf;
typedef __attribute__((ext_vector_type(8)))  float  v8f;

union V16U { v16bf v; v8bf h[2]; };

#define B_      32
#define IDF_    128
#define CDF_    256
#define QL_     16384
#define SL_     24
#define SLP_    32      // padded L
#define QTILE_  128     // queries per block
#define PITCH_  136     // bf16 tile pitch (272B = 17*16B -> 16B aligned rows)

__device__ __forceinline__ bf16_t f2bf(float f) {
    union { float f; unsigned int u; } v; v.f = f;
    unsigned int r = v.u + 0x7FFFu + ((v.u >> 16) & 1u);   // RNE
    unsigned short h = (unsigned short)(r >> 16);
    return __builtin_bit_cast(bf16_t, h);
}

__device__ __forceinline__ v16bf ld_frag(const bf16_t* p0, const bf16_t* p1) {
    V16U u;
    u.h[0] = *reinterpret_cast<const v8bf*>(p0);
    u.h[1] = *reinterpret_cast<const v8bf*>(p1);
    return u.v;
}

// ---------------------------------------------------------------------------
// Kernel A: sourceT[b] = W @ context[b]  -> ws as bf16, layout [b][32][128],
// rows l>=24 zeroed (padding for WMMA K/N).
// grid = (B), block = (128); thread = output channel c.
// ---------------------------------------------------------------------------
__global__ void __launch_bounds__(128)
srcT_kernel(const float* __restrict__ context,   // [B][CDF][SL]
            const float* __restrict__ W,         // [IDF][CDF]
            bf16_t* __restrict__ srcG)           // [B][SLP][IDF]
{
    const int b   = blockIdx.x;
    const int tid = threadIdx.x;

    __shared__ float ctx[CDF_ * SL_];            // 24 KB
    for (int idx = tid; idx < CDF_ * SL_; idx += 128)
        ctx[idx] = context[(size_t)b * CDF_ * SL_ + idx];
    __syncthreads();

    const int c = tid;
    float acc[SL_];
#pragma unroll
    for (int l = 0; l < SL_; ++l) acc[l] = 0.0f;

    const float* wrow = W + (size_t)c * CDF_;
    for (int k = 0; k < CDF_; ++k) {
        const float wv = wrow[k];
        const float* cr = &ctx[k * SL_];
#pragma unroll
        for (int l = 0; l < SL_; ++l) acc[l] += wv * cr[l];
    }

    bf16_t* dst = srcG + (size_t)b * SLP_ * IDF_;
#pragma unroll
    for (int l = 0; l < SL_; ++l) dst[l * IDF_ + c] = f2bf(acc[l]);
#pragma unroll
    for (int l = SL_; l < SLP_; ++l) dst[l * IDF_ + c] = f2bf(0.0f);
}

// ---------------------------------------------------------------------------
// Kernel B: fused scores -> masked softmax -> weighted context, WMMA bf16.
// grid = (QL/QTILE, B), block = 256 threads (8 waves).
// ---------------------------------------------------------------------------
__global__ void __launch_bounds__(256)
attn_kernel(const float* __restrict__ input,     // [B][IDF][QL]
            const bf16_t* __restrict__ srcG,     // [B][SLP][IDF]
            const unsigned char* __restrict__ mask, // [B][SL] bool
            float* __restrict__ wcOut,           // [B][IDF][QL]
            float* __restrict__ attnOut)         // [B][SL][QL]
{
    const int qt  = blockIdx.x;
    const int b   = blockIdx.y;
    const int q0  = qt * QTILE_;
    const int tid = threadIdx.x;
    const int w    = tid >> 5;      // wave 0..7
    const int lane = tid & 31;
    const int m     = lane & 15;
    const int khalf = lane >> 4;

    __shared__ __align__(16) bf16_t tileA[QTILE_][PITCH_];   // 34816 B, [q][c]
    __shared__ __align__(16) bf16_t srcLC[SLP_][IDF_];       //  8192 B, [l][c]
    __shared__ __align__(16) float  scoresL[QTILE_][SLP_];   // 16384 B, [q][l]

    // aliased regions (phase-separated by barriers)
    bf16_t (*pP)[SLP_]    = reinterpret_cast<bf16_t(*)[SLP_]>(&tileA[0][0]);   // probs bf16 [q][l]
    bf16_t (*srcT2)[SLP_] = reinterpret_cast<bf16_t(*)[SLP_]>(&scoresL[0][0]); // srcT bf16 [c][l]

    // ---- Phase 1: async-copy sourceT (8 KB, bf16) global -> LDS via the
    //               CDNA5 async-to-LDS path (ASYNCcnt), then stage the f32
    //               input tile with wide b128 loads, converting to bf16.
    {
        const bf16_t* gsrc = srcG + (size_t)b * SLP_ * IDF_;
        const unsigned ldsbase = (unsigned)(uintptr_t)&srcLC[0][0]; // flat->LDS offset (low 32 bits)
#pragma unroll
        for (int i = 0; i < 2; ++i) {
            const unsigned voff  = (unsigned)(tid + i * 256) * 16u; // 512 x 16B = 8 KB
            const unsigned laddr = ldsbase + voff;
            asm volatile("global_load_async_to_lds_b128 %0, %1, %2 offset:0"
                         :: "v"(laddr), "v"(voff), "s"(gsrc)
                         : "memory");
        }

        // input tile: 128 channels x 128 queries, float4-vectorized over q
        const float4* src4 =
            reinterpret_cast<const float4*>(input + (size_t)b * IDF_ * QL_ + q0);
        for (int idx = tid; idx < IDF_ * (QTILE_ / 4); idx += 256) {
            const int c  = idx >> 5;            // channel
            const int j  = idx & 31;            // float4 index within row
            const float4 v = src4[(size_t)c * (QL_ / 4) + j];
            const int q4 = j * 4;
            tileA[q4 + 0][c] = f2bf(v.x);
            tileA[q4 + 1][c] = f2bf(v.y);
            tileA[q4 + 2][c] = f2bf(v.z);
            tileA[q4 + 3][c] = f2bf(v.w);
        }

        asm volatile("s_wait_asynccnt 0x0" ::: "memory");
    }
    __syncthreads();

    // ---- Phase 2: scores = targetT (128q x 128c) @ srcT (128c x 32L) ----
    {
        v8f acc0 = {}; v8f acc1 = {};
        const int qrow = w * 16 + m;
#pragma unroll
        for (int k = 0; k < 4; ++k) {
            const int cb = k * 32;
            // A: 16x32 bf16, row = query, K = channel (pairs contiguous in LDS)
            v16bf a = ld_frag(&tileA[qrow][cb + khalf * 16],
                              &tileA[qrow][cb + khalf * 16 + 8]);
            // B: 32x16 bf16, row(K) = channel, col(N) = L (lane = N)
            v16bf b0 = ld_frag(&srcLC[m][cb + khalf * 16],
                               &srcLC[m][cb + khalf * 16 + 8]);
            v16bf b1 = ld_frag(&srcLC[16 + m][cb + khalf * 16],
                               &srcLC[16 + m][cb + khalf * 16 + 8]);
            acc0 = __builtin_amdgcn_wmma_f32_16x16x32_bf16(false, a, false, b0,
                                                           (short)0, acc0, false, false);
            acc1 = __builtin_amdgcn_wmma_f32_16x16x32_bf16(false, a, false, b1,
                                                           (short)0, acc1, false, false);
        }
        // C layout: VGPR j -> row m=j+8*khalf, col n=lane&15
#pragma unroll
        for (int j = 0; j < 8; ++j) {
            scoresL[w * 16 + j + 8 * khalf][m]      = acc0[j];
            scoresL[w * 16 + j + 8 * khalf][16 + m] = acc1[j];
        }
    }
    __syncthreads();

    // ---- Phase 3a: masked softmax over L (f32), probs -> bf16 LDS + attnOut ----
    if (tid < QTILE_) {
        const int q = tid;
        const unsigned char* mrow = mask + (size_t)b * SL_;

        unsigned mbits = 0;
#pragma unroll
        for (int l = 0; l < SL_; ++l) mbits |= (mrow[l] ? 1u : 0u) << l;

        float mx = -INFINITY;
#pragma unroll
        for (int l = 0; l < SL_; ++l) {
            const float sv = ((mbits >> l) & 1u) ? -INFINITY : scoresL[q][l];
            mx = fmaxf(mx, sv);
        }
        float sum = 0.0f;
#pragma unroll
        for (int l = 0; l < SL_; ++l) {
            const float sv = ((mbits >> l) & 1u) ? -INFINITY : scoresL[q][l];
            sum += __expf(sv - mx);
        }
        const float inv = 1.0f / sum;
        float* ao = attnOut + ((size_t)b * SL_) * QL_ + q0 + q;
#pragma unroll
        for (int l = 0; l < SL_; ++l) {
            const float sv = ((mbits >> l) & 1u) ? -INFINITY : scoresL[q][l];
            const float p  = __expf(sv - mx) * inv;
            pP[q][l] = f2bf(p);
            ao[(size_t)l * QL_] = p;
        }
#pragma unroll
        for (int l = SL_; l < SLP_; ++l) pP[q][l] = f2bf(0.0f);
    }
    __syncthreads();

    // ---- Phase 3b: transpose srcT into [c][l] (scores region now free) ----
    for (int idx = tid; idx < IDF_ * SLP_; idx += 256) {
        const int c = idx >> 5;
        const int l = idx & 31;
        srcT2[c][l] = srcLC[l][c];
    }
    __syncthreads();

    // ---- Phase 4: wc = srcT (128c x 32L) @ probsT (32L x 128q) ----
    {
        // A: 16x32 bf16, row = channel, K = L
        v16bf a2 = ld_frag(&srcT2[w * 16 + m][khalf * 16],
                           &srcT2[w * 16 + m][khalf * 16 + 8]);
        float* wc = wcOut + (size_t)b * IDF_ * QL_;
#pragma unroll
        for (int nt = 0; nt < 8; ++nt) {
            // B: 32x16 bf16, row(K) = L, col(N) = query (lane = N)
            v16bf bb = ld_frag(&pP[nt * 16 + m][khalf * 16],
                               &pP[nt * 16 + m][khalf * 16 + 8]);
            v8f c2 = {};
            c2 = __builtin_amdgcn_wmma_f32_16x16x32_bf16(false, a2, false, bb,
                                                         (short)0, c2, false, false);
#pragma unroll
            for (int j = 0; j < 8; ++j) {
                const int c = w * 16 + j + 8 * khalf;
                wc[(size_t)c * QL_ + q0 + nt * 16 + m] = c2[j];
            }
        }
    }
}

// ---------------------------------------------------------------------------
extern "C" void kernel_launch(void* const* d_in, const int* in_sizes, int n_in,
                              void* d_out, int out_size, void* d_ws, size_t ws_size,
                              hipStream_t stream) {
    const float*         inp  = (const float*)d_in[0];          // (32,128,128,128)
    const float*         ctx  = (const float*)d_in[1];          // (32,256,24)
    const float*         Wm   = (const float*)d_in[2];          // (128,256)
    const unsigned char* mask = (const unsigned char*)d_in[3];  // (32,24) bool

    float* wcOut   = (float*)d_out;                                   // (32,128,128,128)
    float* attnOut = wcOut + (size_t)B_ * IDF_ * QL_;                 // (32,24,128,128)

    bf16_t* srcG = (bf16_t*)d_ws;   // [B][32][128] bf16 = 256 KB

    srcT_kernel<<<dim3(B_), dim3(128), 0, stream>>>(ctx, Wm, srcG);
    attn_kernel<<<dim3(QL_ / QTILE_, B_), dim3(256), 0, stream>>>(
        inp, srcG, mask, wcOut, attnOut);
}
